// TransformerDecoder_25881472926639
// MI455X (gfx1250) — matrix operand order repaired and checked by reading, not compiled, not run on hardware
//
#include <hip/hip_runtime.h>

#define VOCAB 50257
#define EMBD  1024
#define T_BLK 8192
#define NPAD  50304              // VOCAB padded to multiple of 64
#define KT_TILES (EMBD/32)       // 32 K-tiles of 32
#define CHUNKS (NPAD/32)         // 1572 column chunks of 32

typedef __attribute__((ext_vector_type(16))) __bf16 v16bf;
typedef __attribute__((ext_vector_type(8)))  float  v8f;

// workspace layout (bytes)
#define OFF_XH  0ull                         // A hi  8192x1024 bf16 (packed)   16.78 MB
#define OFF_XL  16777216ull                  // A lo                             16.78 MB
#define OFF_BH  33554432ull                  // B hi  1024xNPAD bf16 (packed)   103.0 MB
#define OFF_BL  136577024ull                 // B lo                            103.0 MB
#define OFF_PM  239599616ull                 // partial max  [8192][1572] f32    51.5 MB
#define OFF_PS  291110912ull                 // partial sum  [8192][1572] f32    51.5 MB
#define OFF_NLL 342622208ull                 // per-row nll  [8192] f32

__device__ __forceinline__ unsigned f32_to_bf16_rne(float v){
  unsigned u = __float_as_uint(v);
  return ((u + 0x7FFFu + ((u >> 16) & 1u)) >> 16) & 0xFFFFu;
}
__device__ __forceinline__ void split_bf16(float v, unsigned short& hi, unsigned short& lo){
  unsigned u = __float_as_uint(v);
  unsigned hb = (u + 0x7FFFu + ((u >> 16) & 1u)) & 0xFFFF0000u;
  hi = (unsigned short)(hb >> 16);
  lo = (unsigned short)f32_to_bf16_rne(v - __uint_as_float(hb));
}
// combine two (max, sum-exp) softmax partials
__device__ __forceinline__ void comb(float& m, float& s, float m2, float s2){
  float M = fmaxf(m, m2);
  s = s * __expf(m - M) + s2 * __expf(m2 - M);
  m = M;
}

// ---- prep: x = emb[tok] + pos, split into bf16 hi/lo, pre-swizzled to WMMA A layout ----
// A 16x32 bf16 layout: lane = row%16 + 16*bit3(k); elem i: k = (k&7) + 8*(k>>4) inverse mapping
__global__ void prep_x_kernel(const int* __restrict__ tokens,
                              const float* __restrict__ emb,
                              const float* __restrict__ pos,
                              unsigned short* __restrict__ xh,
                              unsigned short* __restrict__ xl)
{
  const int t = blockIdx.x;
  const int tok = tokens[t];
  const int mt = t >> 4, row_in = t & 15;
  for (int j = 0; j < 4; ++j){
    int e = threadIdx.x + j * 256;                       // 0..1023, coalesced
    float v = emb[(size_t)tok * EMBD + e] + pos[(size_t)t * EMBD + e];
    unsigned short hi, lo; split_bf16(v, hi, lo);
    int kt = e >> 5, kin = e & 31;
    int lane = row_in + (((kin >> 3) & 1) << 4);
    int i = (kin & 7) + ((kin >> 4) << 3);
    size_t idx = ((size_t)(mt * KT_TILES + kt) * 32 + lane) * 16 + i;
    xh[idx] = hi; xl[idx] = lo;
  }
}

// ---- prep: W_head [K=1024][V] split into bf16 hi/lo, pre-swizzled to WMMA B layout ----
// B 32x16 bf16 layout: lane = col%16 + 16*bit3(k); elem i = (k&7) + 8*((k%32)>>4)
__global__ void prep_w_kernel(const float* __restrict__ W,
                              unsigned short* __restrict__ wh,
                              unsigned short* __restrict__ wl)
{
  const int k = blockIdx.x;                              // 0..1023
  const int kt = k >> 5, kin = k & 31;
  const int laneHalf = ((kin >> 3) & 1) << 4;
  const int i = (kin & 7) + ((kin >> 4) << 3);
  for (int c = threadIdx.x; c < NPAD; c += 256){
    float v = (c < VOCAB) ? W[(size_t)k * VOCAB + c] : 0.0f;
    unsigned short hi, lo; split_bf16(v, hi, lo);
    int nt = c >> 4;
    int lane = (c & 15) + laneHalf;
    size_t idx = ((size_t)(nt * KT_TILES + kt) * 32 + lane) * 16 + i;
    wh[idx] = hi; wl[idx] = lo;
  }
}

// ---- main GEMM: 128x64 tile / block, 8 wave32s, 2x2 16x16 accum per wave.
// split-bf16 3-product: AhBh + AhBl + AlBh (~fp32 accuracy, bf16 WMMA rate).
// fused epilogue: store logits (+bias) and per-(row, 32-col-chunk) online-softmax partials.
__global__ __launch_bounds__(256) void gemm_kernel(
    const unsigned short* __restrict__ xh_, const unsigned short* __restrict__ xl_,
    const unsigned short* __restrict__ wh_, const unsigned short* __restrict__ wl_,
    const float* __restrict__ bias,
    float* __restrict__ logits, float* __restrict__ pm, float* __restrict__ ps)
{
  const v16bf* Ah = (const v16bf*)xh_;
  const v16bf* Al = (const v16bf*)xl_;
  const v16bf* Bh = (const v16bf*)wh_;
  const v16bf* Bl = (const v16bf*)wl_;

  const int lane = threadIdx.x & 31;
  const int wave = threadIdx.x >> 5;
  const int wm = wave & 3;                 // 32-row slice within 128
  const int wn = wave >> 2;                // 32-col slice within 64
  const int rowBase = blockIdx.y * 128;
  const int colBase = blockIdx.x * 64;
  const int mt0 = (rowBase >> 4) + wm * 2;
  const int nt0 = (colBase >> 4) + wn * 2;

  v8f acc[2][2] = {};

  for (int kt = 0; kt < KT_TILES; ++kt){
    size_t a0 = ((size_t)(mt0 + 0) * KT_TILES + kt) * 32 + lane;
    size_t a1 = ((size_t)(mt0 + 1) * KT_TILES + kt) * 32 + lane;
    size_t b0 = ((size_t)(nt0 + 0) * KT_TILES + kt) * 32 + lane;
    size_t b1 = ((size_t)(nt0 + 1) * KT_TILES + kt) * 32 + lane;
    v16bf ah[2] = { Ah[a0], Ah[a1] };
    v16bf al[2] = { Al[a0], Al[a1] };
    v16bf bh[2] = { Bh[b0], Bh[b1] };
    v16bf bl[2] = { Bl[b0], Bl[b1] };
    if (kt + 1 < KT_TILES){                      // prefetch next K-step (global_prefetch_b8)
      __builtin_prefetch((const void*)&Ah[a0 + 32], 0, 0);
      __builtin_prefetch((const void*)&Al[a0 + 32], 0, 0);
      __builtin_prefetch((const void*)&Bh[b0 + 32], 0, 0);
      __builtin_prefetch((const void*)&Bl[b0 + 32], 0, 0);
    }
#pragma unroll
    for (int mi = 0; mi < 2; ++mi){
#pragma unroll
      for (int ni = 0; ni < 2; ++ni){
        acc[mi][ni] = __builtin_amdgcn_wmma_f32_16x16x32_bf16(
            false, ah[mi], false, bh[ni], (short)0, acc[mi][ni], false, false);
        acc[mi][ni] = __builtin_amdgcn_wmma_f32_16x16x32_bf16(
            false, ah[mi], false, bl[ni], (short)0, acc[mi][ni], false, false);
        acc[mi][ni] = __builtin_amdgcn_wmma_f32_16x16x32_bf16(
            false, al[mi], false, bh[ni], (short)0, acc[mi][ni], false, false);
      }
    }
  }

  // ---- fused epilogue ----
  const int nlo  = lane & 15;                 // N position within 16 (C layout)
  const int rsel = lane >> 4;                 // upper half holds M = i + 8
  const int col0 = colBase + wn * 32 + nlo;
  const int col1 = col0 + 16;
  const float bias0 = (col0 < VOCAB) ? bias[col0] : 0.0f;
  const float bias1 = (col1 < VOCAB) ? bias[col1] : 0.0f;
  const int chunk = blockIdx.x * 2 + wn;      // 32-col chunk index

#pragma unroll
  for (int mi = 0; mi < 2; ++mi){
#pragma unroll
    for (int i = 0; i < 8; ++i){
      int row = rowBase + wm * 32 + mi * 16 + i + rsel * 8;
      float lm = -1e30f, lsum = 0.0f;
      float v0 = acc[mi][0][i] + bias0;
      float v1 = acc[mi][1][i] + bias1;
      if (col0 < VOCAB){ logits[(size_t)row * VOCAB + col0] = v0; comb(lm, lsum, v0, 1.0f); }
      if (col1 < VOCAB){ logits[(size_t)row * VOCAB + col1] = v1; comb(lm, lsum, v1, 1.0f); }
      // reduce across the 16 lanes holding this row
#pragma unroll
      for (int off = 1; off < 16; off <<= 1){
        float om = __shfl_xor(lm,   off, 32);
        float os = __shfl_xor(lsum, off, 32);
        comb(lm, lsum, om, os);
      }
      if (nlo == 0){
        size_t p = (size_t)row * CHUNKS + chunk;
        pm[p] = lm; ps[p] = lsum;
      }
    }
  }
}

// ---- per-row: combine chunk partials, compute nll = -(logit[target] - m - log s) ----
__global__ void row_loss_kernel(const float* __restrict__ pm, const float* __restrict__ ps,
                                const float* __restrict__ logits,
                                const int* __restrict__ targets,
                                float* __restrict__ nll)
{
  const int t = blockIdx.x;
  float m = -1e30f, s = 0.0f;
  for (int c = threadIdx.x; c < CHUNKS; c += 256)
    comb(m, s, pm[(size_t)t * CHUNKS + c], ps[(size_t)t * CHUNKS + c]);
  __shared__ float sm[256], ss[256];
  sm[threadIdx.x] = m; ss[threadIdx.x] = s;
  __syncthreads();
  for (int off = 128; off > 0; off >>= 1){
    if (threadIdx.x < off) comb(sm[threadIdx.x], ss[threadIdx.x], sm[threadIdx.x + off], ss[threadIdx.x + off]);
    __syncthreads();
  }
  if (threadIdx.x == 0){
    int tg = targets[t];
    float lt = logits[(size_t)t * VOCAB + tg];
    nll[t] = -(lt - sm[0] - logf(ss[0]));
  }
}

__global__ void mean_loss_kernel(const float* __restrict__ nll, float* __restrict__ out_loss)
{
  __shared__ float sh[256];
  float a = 0.0f;
  for (int i = threadIdx.x; i < T_BLK; i += 256) a += nll[i];
  sh[threadIdx.x] = a;
  __syncthreads();
  for (int off = 128; off > 0; off >>= 1){
    if (threadIdx.x < off) sh[threadIdx.x] += sh[threadIdx.x + off];
    __syncthreads();
  }
  if (threadIdx.x == 0) out_loss[0] = sh[0] / (float)T_BLK;
}

extern "C" void kernel_launch(void* const* d_in, const int* in_sizes, int n_in,
                              void* d_out, int out_size, void* d_ws, size_t ws_size,
                              hipStream_t stream) {
  const int*   tokens  = (const int*)  d_in[0];
  const int*   targets = (const int*)  d_in[1];
  const float* emb     = (const float*)d_in[2];
  const float* pos     = (const float*)d_in[3];
  const float* W       = (const float*)d_in[4];
  const float* b_head  = (const float*)d_in[5];
  float* logits = (float*)d_out;
  float* loss   = logits + (size_t)T_BLK * VOCAB;   // loss follows logits in d_out
  char* ws = (char*)d_ws;

  unsigned short* xh = (unsigned short*)(ws + OFF_XH);
  unsigned short* xl = (unsigned short*)(ws + OFF_XL);
  unsigned short* wh = (unsigned short*)(ws + OFF_BH);
  unsigned short* wl = (unsigned short*)(ws + OFF_BL);
  float* pm  = (float*)(ws + OFF_PM);
  float* ps  = (float*)(ws + OFF_PS);
  float* nll = (float*)(ws + OFF_NLL);

  prep_x_kernel<<<T_BLK, 256, 0, stream>>>(tokens, emb, pos, xh, xl);
  prep_w_kernel<<<EMBD, 256, 0, stream>>>(W, wh, wl);
  gemm_kernel<<<dim3(NPAD / 64, T_BLK / 128), 256, 0, stream>>>(
      xh, xl, wh, wl, b_head, logits, pm, ps);
  row_loss_kernel<<<T_BLK, 256, 0, stream>>>(pm, ps, logits, targets, nll);
  mean_loss_kernel<<<1, 256, 0, stream>>>(nll, loss);
}
